// MixtureOfGranularities_40355512714066
// MI455X (gfx1250) — matrix operand order repaired and checked by reading, not compiled, not run on hardware
//
#include <hip/hip_runtime.h>
#include <hip/hip_bf16.h>

// ---------- problem constants ----------
#define DD   768
#define FF   2048
#define SS   3
#define EPC  4
#define EE   12
#define BT   2048
#define TOPK 2

// expert-kernel tiling
#define TM 32    // tokens per workgroup
#define FT 128   // f-chunk
#define KB 64    // k-block for gemm1/2
#define NB 128   // n-block for gemm3

typedef __attribute__((ext_vector_type(16))) __bf16 v16bf;
typedef __attribute__((ext_vector_type(8)))  __bf16 v8bf;
typedef __attribute__((ext_vector_type(8)))  float  v8f;

__device__ __forceinline__ __bf16 f2bf(float f) {
  unsigned u = __builtin_bit_cast(unsigned, f);
  unsigned r = (u + 0x7FFFu + ((u >> 16) & 1u)) >> 16;
  unsigned short s = (unsigned short)r;
  return __builtin_bit_cast(__bf16, s);
}

__device__ __forceinline__ v8f zero8() {
  v8f z = {0.f,0.f,0.f,0.f,0.f,0.f,0.f,0.f};
  return z;
}

// A fragment, 16x32 bf16: lane<16 -> K k0..k0+15,
// lane>=16 -> K k0+16..k0+31; M = m_base + (lane&15).
__device__ __forceinline__ v16bf load_a16(const __bf16* base, int m_base, int ld,
                                          int k0, int hi, int l15) {
  const __bf16* p = base + (size_t)(m_base + l15) * ld + k0 + hi * 16;
  v8bf lo = *(const v8bf*)p;
  v8bf hh = *(const v8bf*)(p + 8);
  v16bf v;
#pragma unroll
  for (int i = 0; i < 8; ++i) { v[i] = lo[i]; v[i + 8] = hh[i]; }
  return v;
}

// B fragment, 32x16 bf16 from N-major (transposed) LDS: lane<16 -> K k0..k0+15,
// lane>=16 -> K k0+16..k0+31; N = n_base + (lane&15). 16 contiguous bf16.
__device__ __forceinline__ v16bf load_b16(const __bf16* baseT, int n_base, int ld,
                                          int k0, int hi, int l15) {
  const __bf16* p = baseT + (size_t)(n_base + l15) * ld + k0 + hi * 16;
  return *(const v16bf*)p;
}

__device__ __forceinline__ v8f wmma_bf16(v16bf a, v16bf b, v8f c) {
  return __builtin_amdgcn_wmma_f32_16x16x32_bf16(false, a, false, b,
                                                 (short)0, c, false, false);
}

// ---------------- init ----------------
__global__ __launch_bounds__(256) void init_kernel(float* __restrict__ out,
                                                   float* __restrict__ stats) {
  size_t i = (size_t)blockIdx.x * blockDim.x + threadIdx.x;
  size_t n = (size_t)BT * DD;
  for (size_t idx = i; idx < n; idx += (size_t)gridDim.x * blockDim.x) out[idx] = 0.f;
  if (i < 2 * EE) stats[i] = 0.f;
}

// ---------------- router: softmax + top2 + combine + aux stats ----------------
__global__ __launch_bounds__(256) void router_kernel(const float* __restrict__ x,
                                                     const float* __restrict__ rw,
                                                     float* __restrict__ combine,
                                                     float* __restrict__ stats) {
  __shared__ float red[256];
  __shared__ float gate[EE];
  __shared__ int   topi[2];
  __shared__ float topw[2];
  const int t = blockIdx.x;
  const int tid = threadIdx.x;
  for (int e = 0; e < EE; ++e) {
    float p = 0.f;
    for (int d = tid; d < DD; d += 256) p += x[(size_t)t * DD + d] * rw[d * EE + e];
    red[tid] = p; __syncthreads();
    for (int off = 128; off > 0; off >>= 1) {
      if (tid < off) red[tid] += red[tid + off];
      __syncthreads();
    }
    if (tid == 0) gate[e] = red[0];
    __syncthreads();
  }
  if (tid == 0) {
    float mx = gate[0];
    for (int e = 1; e < EE; ++e) mx = fmaxf(mx, gate[e]);
    float sum = 0.f;
    for (int e = 0; e < EE; ++e) { gate[e] = expf(gate[e] - mx); sum += gate[e]; }
    float inv = 1.f / sum;
    for (int e = 0; e < EE; ++e) gate[e] *= inv;
    int i1 = 0;
    for (int e = 1; e < EE; ++e) if (gate[e] > gate[i1]) i1 = e;
    int i2 = (i1 == 0) ? 1 : 0;
    for (int e = 0; e < EE; ++e) if (e != i1 && gate[e] > gate[i2]) i2 = e;
    float p1 = gate[i1], p2 = gate[i2];
    float tw = p1 + p2 + 1e-8f;
    topi[0] = i1; topi[1] = i2; topw[0] = p1 / tw; topw[1] = p2 / tw;
    atomicAdd(&stats[i1], 1.f);
    atomicAdd(&stats[i2], 1.f);
  }
  __syncthreads();
  if (tid < EE) {
    float c = (tid == topi[0]) ? topw[0] : ((tid == topi[1]) ? topw[1] : 0.f);
    combine[(size_t)t * EE + tid] = c;
    atomicAdd(&stats[EE + tid], gate[tid]);
  }
}

__global__ void finalize_aux_kernel(const float* __restrict__ stats,
                                    float* __restrict__ out_aux) {
  float aux = 0.f;
  for (int e = 0; e < EE; ++e) {
    float frac = stats[e] / ((float)BT * (float)TOPK);
    float load = stats[EE + e] / (float)BT;
    aux += frac * load;
  }
  *out_aux = (float)EE * aux;
}

// ---------------- f32 -> bf16 convert (scale-0 repr) ----------------
__global__ __launch_bounds__(256) void cvt_bf16_kernel(const float* __restrict__ in,
                                                       __bf16* __restrict__ o, int n) {
  int i = blockIdx.x * blockDim.x + threadIdx.x;
  if (i < n) o[i] = f2bf(in[i]);
}

// ---------------- pooled repr: (M x KP) @ (KP x 768) -> repeated bf16 ----------------
template<int KP, int PS>
__global__ __launch_bounds__(256) void pool_gemm_kernel(const float* __restrict__ x,
                                                        const float* __restrict__ w,
                                                        __bf16* __restrict__ rout) {
  __shared__ __bf16 As[32 * 64];    //  4 KB, row-major [m][k]
  __shared__ __bf16 WT[128 * 64];   // 16 KB, N-major   [n][k]
  const int tid = threadIdx.x, wv = tid >> 5, lane = tid & 31;
  const int hi = lane >> 4, l15 = lane & 15;
  const int g0 = blockIdx.x * 32;
  const int n0 = blockIdx.y * 128;
  v8f acc0 = zero8(), acc1 = zero8();
  for (int k0 = 0; k0 < KP; k0 += 64) {
    __syncthreads();
    for (int i = tid; i < 32 * 64; i += 256) {
      int kk = i & 63, m = i >> 6;
      As[m * 64 + kk] = f2bf(x[(size_t)(g0 + m) * KP + k0 + kk]);
    }
    for (int i = tid; i < 128 * 64; i += 256) {
      int n = i & 127, kk = i >> 7;
      WT[n * 64 + kk] = f2bf(w[(size_t)(k0 + kk) * DD + n0 + n]);
    }
    __syncthreads();
#pragma unroll
    for (int kk = 0; kk < 64; kk += 32) {
      v16bf a0 = load_a16(As, 0, 64, kk, hi, l15);
      v16bf a1 = load_a16(As, 16, 64, kk, hi, l15);
      v16bf b  = load_b16(WT, wv * 16, 64, kk, hi, l15);
      acc0 = wmma_bf16(a0, b, acc0);
      acc1 = wmma_bf16(a1, b, acc1);
    }
  }
  const int col = n0 + wv * 16 + l15;
#pragma unroll
  for (int j = 0; j < 8; ++j) {
    int m0v = j + hi * 8;
    int m1v = 16 + j + hi * 8;
    __bf16 v0 = f2bf(acc0[j]);
    __bf16 v1 = f2bf(acc1[j]);
    for (int rp = 0; rp < PS; ++rp) {
      rout[((size_t)(g0 + m0v) * PS + rp) * DD + col] = v0;
      rout[((size_t)(g0 + m1v) * PS + rp) * DD + col] = v1;
    }
  }
}

// ---------------- fused expert FFN: g/u GEMM -> silu*u -> down GEMM -> atomic out ----------------
__global__ __launch_bounds__(256) void expert_ffn_kernel(
    const __bf16* __restrict__ r0, const __bf16* __restrict__ r1,
    const __bf16* __restrict__ r2,
    const float* __restrict__ wg, const float* __restrict__ wu,
    const float* __restrict__ wd,
    const float* __restrict__ combine, float* __restrict__ out) {
  extern __shared__ char smem[];
  __bf16* rt   = (__bf16*)smem;                                // [TM][DD]  48 KB
  float*  eo   = (float*)(smem + TM * DD * 2);                 // [TM][DD]  96 KB
  __bf16* hbuf = (__bf16*)(smem + TM * DD * 2 + TM * DD * 4);  // [TM][FT]   8 KB
  __bf16* stg  = (__bf16*)(smem + TM * DD * 6 + TM * FT * 2);  //           32 KB
  __bf16* wgT = stg;              // [FT][KB]
  __bf16* wuT = stg + FT * KB;    // [FT][KB]
  __bf16* wdT = stg;              // [NB][NB] (reused, barrier-protected)

  const int tid = threadIdx.x;
  const int wv = tid >> 5, lane = tid & 31;
  const int hi = lane >> 4, l15 = lane & 15;

  const int s  = blockIdx.z;
  const int e  = blockIdx.y;
  const int se = s * EPC + e;
  const int t0 = blockIdx.x * TM;

  const __bf16* r  = (s == 0) ? r0 : (s == 1) ? r1 : r2;
  const float* wgp = wg + (size_t)se * DD * FF;
  const float* wup = wu + (size_t)se * DD * FF;
  const float* wdp = wd + (size_t)se * FF * DD;

  for (int i = tid; i < TM * DD; i += 256) {
    rt[i] = r[(size_t)t0 * DD + i];
    eo[i] = 0.f;
  }
  __syncthreads();

  for (int f0 = 0; f0 < FF; f0 += FT) {
    // ---- GEMM1/2: g,u = r @ wg/wu  (32 x FT), K = 768 ----
    v8f ag0 = zero8(), ag1 = zero8(), au0 = zero8(), au1 = zero8();
    for (int k0 = 0; k0 < DD; k0 += KB) {
      __syncthreads();
      for (int i = tid; i < FT * KB; i += 256) {
        int n = i & (FT - 1), kk = i >> 7;
        wgT[n * KB + kk] = f2bf(wgp[(size_t)(k0 + kk) * FF + f0 + n]);
        wuT[n * KB + kk] = f2bf(wup[(size_t)(k0 + kk) * FF + f0 + n]);
      }
      if (k0 + KB < DD)  // hint next weight k-block into cache (global_prefetch_b8)
        __builtin_prefetch(&wgp[(size_t)(k0 + KB) * FF + f0 + tid], 0, 1);
      __syncthreads();
#pragma unroll
      for (int kk = 0; kk < KB; kk += 32) {
        v16bf a0 = load_a16(rt, 0,  DD, k0 + kk, hi, l15);
        v16bf a1 = load_a16(rt, 16, DD, k0 + kk, hi, l15);
        v16bf bg = load_b16(wgT, wv * 16, KB, kk, hi, l15);
        v16bf bu = load_b16(wuT, wv * 16, KB, kk, hi, l15);
        ag0 = wmma_bf16(a0, bg, ag0);
        ag1 = wmma_bf16(a1, bg, ag1);
        au0 = wmma_bf16(a0, bu, au0);
        au1 = wmma_bf16(a1, bu, au1);
      }
    }
    // ---- h = silu(g) * u -> LDS (bf16) ----
#pragma unroll
    for (int j = 0; j < 8; ++j) {
      float g0v = ag0[j], u0v = au0[j];
      float g1v = ag1[j], u1v = au1[j];
      float h0 = (g0v / (1.f + expf(-g0v))) * u0v;
      float h1 = (g1v / (1.f + expf(-g1v))) * u1v;
      hbuf[(j + hi * 8) * FT + wv * 16 + l15]        = f2bf(h0);
      hbuf[(16 + j + hi * 8) * FT + wv * 16 + l15]   = f2bf(h1);
    }
    __syncthreads();
    // ---- GEMM3 partial: eo += h(32 x FT) @ wd[f0:f0+FT, :] ----
    for (int n0 = 0; n0 < DD; n0 += NB) {
      __syncthreads();
      for (int i = tid; i < NB * NB; i += 256) {
        int n = i & (NB - 1), kk = i >> 7;
        wdT[n * NB + kk] = f2bf(wdp[(size_t)(f0 + kk) * DD + n0 + n]);
      }
      __syncthreads();
#pragma unroll
      for (int m_t = 0; m_t < 2; ++m_t) {
        v8f c;
#pragma unroll
        for (int j = 0; j < 8; ++j)
          c[j] = eo[(m_t * 16 + j + hi * 8) * DD + n0 + wv * 16 + l15];
#pragma unroll
        for (int kk = 0; kk < NB; kk += 32) {
          v16bf a = load_a16(hbuf, m_t * 16, FT, kk, hi, l15);
          v16bf b = load_b16(wdT, wv * 16, NB, kk, hi, l15);
          c = wmma_bf16(a, b, c);
        }
#pragma unroll
        for (int j = 0; j < 8; ++j)
          eo[(m_t * 16 + j + hi * 8) * DD + n0 + wv * 16 + l15] = c[j];
      }
    }
    __syncthreads();
  }
  // ---- epilogue: out[t,:] += combine[t, se] * eo  (skip unselected experts) ----
  for (int i = tid; i < TM * DD; i += 256) {
    int m = i / DD, d = i - m * DD;
    float w = combine[(size_t)(t0 + m) * EE + se];
    if (w != 0.f) atomicAdd(&out[(size_t)(t0 + m) * DD + d], w * eo[i]);
  }
}

extern "C" void kernel_launch(void* const* d_in, const int* in_sizes, int n_in,
                              void* d_out, int out_size, void* d_ws, size_t ws_size,
                              hipStream_t stream) {
  (void)in_sizes; (void)n_in; (void)out_size; (void)ws_size;
  const float* x        = (const float*)d_in[0];
  const float* router_w = (const float*)d_in[1];
  const float* proj_w4  = (const float*)d_in[2];
  const float* proj_w16 = (const float*)d_in[3];
  const float* wg       = (const float*)d_in[4];
  const float* wu       = (const float*)d_in[5];
  const float* wd       = (const float*)d_in[6];
  float* out = (float*)d_out;

  char* ws = (char*)d_ws;
  float*  combine = (float*)ws;                                   // BT*EE f32
  float*  stats   = (float*)(ws + (size_t)BT * EE * 4);           // 24 f32 (pad 256)
  __bf16* r0      = (__bf16*)(ws + (size_t)BT * EE * 4 + 256);
  __bf16* r1      = r0 + (size_t)BT * DD;
  __bf16* r2      = r1 + (size_t)BT * DD;

  init_kernel<<<256, 256, 0, stream>>>(out, stats);
  router_kernel<<<BT, 256, 0, stream>>>(x, router_w, combine, stats);
  cvt_bf16_kernel<<<(BT * DD + 255) / 256, 256, 0, stream>>>(x, r0, BT * DD);
  pool_gemm_kernel<4 * DD, 4><<<dim3((BT / 4) / 32, DD / 128), 256, 0, stream>>>(x, proj_w4, r1);
  pool_gemm_kernel<16 * DD, 16><<<dim3((BT / 16) / 32, DD / 128), 256, 0, stream>>>(x, proj_w16, r2);
  finalize_aux_kernel<<<1, 1, 0, stream>>>(stats, out + (size_t)BT * DD);

  dim3 grid(BT / TM, EPC, SS);
  size_t smem = (size_t)TM * DD * 2 + (size_t)TM * DD * 4 + (size_t)TM * FT * 2 + 32768;
  expert_ffn_kernel<<<grid, 256, smem, stream>>>(r0, r1, r2, wg, wu, wd, combine, out);
}
